// Attention2_7748121002524
// MI455X (gfx1250) — matrix operand order repaired and checked by reading, not compiled, not run on hardware
//
#include <hip/hip_runtime.h>
#include <hip/hip_bf16.h>

// ---------------- problem constants ----------------
constexpr int Bb = 4, Nn = 1024, Cc = 768, Hh = 12, Dd = 64;
constexpr float SCALE = 0.125f;                    // D^-0.5
constexpr size_t BNC = (size_t)Bb * Nn * Cc;       // 3,145,728 (= B*H*N*D too)

typedef __attribute__((ext_vector_type(16))) __bf16 v16bf;
typedef __attribute__((ext_vector_type(8)))  __bf16 v8bf;
typedef __attribute__((ext_vector_type(8)))  float  v8f;

static __device__ __forceinline__ __bf16 f2bf(float f) {
    unsigned u = __builtin_bit_cast(unsigned, f);
    unsigned r = (u + 0x7FFFu + ((u >> 16) & 1u)) >> 16;   // round-nearest-even
    unsigned short s = (unsigned short)r;
    return __builtin_bit_cast(__bf16, s);
}

static __device__ __forceinline__ v16bf cat8(v8bf lo, v8bf hi) {
    return __builtin_shufflevector(lo, hi, 0,1,2,3,4,5,6,7,8,9,10,11,12,13,14,15);
}

static __device__ __forceinline__ v8f wmma_bf16(v16bf a, v16bf b, v8f c) {
    return __builtin_amdgcn_wmma_f32_16x16x32_bf16(false, a, false, b, (short)0, c, false, false);
}

// ---------------- kernel 0: f32 -> bf16 staging ----------------
__global__ void cvt_kernel(const float* __restrict__ src, __bf16* __restrict__ dst, int n) {
    int i = blockIdx.x * blockDim.x + threadIdx.x;
    int stride = gridDim.x * blockDim.x;
    for (; i < n; i += stride) dst[i] = f2bf(src[i]);
}

// ---------------- kernel 1: QKV GEMM + bias, scatter to Q/K/Vt ----------------
// 32x32 tile per wave (2x2 register-blocked WMMA).
// xb: [2][B*N][C] bf16, wq: [3C][C] bf16, bias: [3C] f32
// Qo/Ko: [2][B,H,N,D] bf16 (Q pre-scaled), Vt: [2][B,H,D,N] bf16
__global__ void __launch_bounds__(128) qkv_kernel(
    const __bf16* __restrict__ xb, const __bf16* __restrict__ wq,
    const float* __restrict__ bias,
    __bf16* __restrict__ Qo, __bf16* __restrict__ Ko, __bf16* __restrict__ Vt)
{
    constexpr int NT = (3 * Cc) / 32;   // 72
    constexpr int MT = (Bb * Nn) / 32;  // 128
    const int lane = threadIdx.x & 31;
    const int l16 = lane & 15, hv = lane >> 4;
    int wid = blockIdx.x * 4 + (threadIdx.x >> 5);
    const int s  = wid / (MT * NT); wid -= s * (MT * NT);
    const int mt = wid / NT;
    const int nt = wid - mt * NT;

    const __bf16* xrow0 = xb + (size_t)s * BNC + (size_t)(mt * 32 + l16) * Cc;
    const __bf16* xrow1 = xrow0 + (size_t)16 * Cc;
    const __bf16* wrow0 = wq + (size_t)(nt * 32 + l16) * Cc + 16 * hv;
    const __bf16* wrow1 = wrow0 + (size_t)16 * Cc;

    v8f acc[2][2];
    acc[0][0] = (v8f){}; acc[0][1] = (v8f){}; acc[1][0] = (v8f){}; acc[1][1] = (v8f){};
    #pragma unroll 2
    for (int k0 = 0; k0 < Cc; k0 += 32) {
        v16bf a0 = *(const v16bf*)(xrow0 + k0 + 16 * hv);
        v16bf a1 = *(const v16bf*)(xrow1 + k0 + 16 * hv);
        v16bf b0 = *(const v16bf*)(wrow0 + k0);
        v16bf b1 = *(const v16bf*)(wrow1 + k0);
        acc[0][0] = wmma_bf16(a0, b0, acc[0][0]);
        acc[0][1] = wmma_bf16(a0, b1, acc[0][1]);
        acc[1][0] = wmma_bf16(a1, b0, acc[1][0]);
        acc[1][1] = wmma_bf16(a1, b1, acc[1][1]);
    }

    const size_t so = (size_t)s * BNC;
    #pragma unroll
    for (int ns = 0; ns < 2; ns++) {
        const int n = nt * 32 + ns * 16 + l16;
        const int which = n / Cc, cin = n % Cc;
        const int h = cin / Dd, d = cin % Dd;
        const float bval = bias[n];
        #pragma unroll
        for (int ms = 0; ms < 2; ms++) {
            #pragma unroll
            for (int r = 0; r < 8; r++) {
                const int m  = mt * 32 + ms * 16 + r + 8 * hv;
                const int bb = m >> 10, nn = m & (Nn - 1);
                float v = acc[ms][ns][r] + bval;
                if (which == 0) v *= SCALE;
                __bf16 bv = f2bf(v);
                if (which == 0)      Qo[so + ((size_t)(bb * Hh + h) * Nn + nn) * Dd + d] = bv;
                else if (which == 1) Ko[so + ((size_t)(bb * Hh + h) * Nn + nn) * Dd + d] = bv;
                else                 Vt[so + ((size_t)(bb * Hh + h) * Dd + d) * Nn + nn] = bv;
            }
        }
    }
}

// ---------------- kernel 2: dual-stream flash attention ----------------
// Q/K: [2][B,H,N,D], Vt: [2][B,H,D,N], O: [2][B*N][C] bf16
__global__ void __launch_bounds__(128) attn_kernel(
    const __bf16* __restrict__ Q, const __bf16* __restrict__ Kd,
    const __bf16* __restrict__ Vt, __bf16* __restrict__ O)
{
    __shared__ __align__(64) __bf16 pbuf[4][16][32];  // per-wave P staging
    const int w = threadIdx.x >> 5;
    const int lane = threadIdx.x & 31;
    const int l16 = lane & 15, hv = lane >> 4;
    const int wid = blockIdx.x * 4 + w;
    constexpr int QT = Nn / 16;                 // 64
    const int bh = wid / QT, qt = wid % QT;     // bh in [0,48)
    const size_t SQ = BNC;                      // per-stream stride
    const size_t qkbase = (size_t)bh * Nn * Dd;

    // preload Q A-fragments: [stream][k0-chunk]
    v16bf aq[2][2];
    #pragma unroll
    for (int s2 = 0; s2 < 2; s2++) {
        const __bf16* qrow = Q + s2 * SQ + qkbase + (size_t)(qt * 16 + l16) * Dd;
        #pragma unroll
        for (int kc = 0; kc < 2; kc++) {
            aq[s2][kc] = *(const v16bf*)(qrow + kc * 32 + 16 * hv);
        }
    }

    float mrow[8], lrow[8];
    #pragma unroll
    for (int r = 0; r < 8; r++) { mrow[r] = -INFINITY; lrow[r] = 0.f; }
    v8f oacc[2][4];
    #pragma unroll
    for (int s2 = 0; s2 < 2; s2++)
        #pragma unroll
        for (int t = 0; t < 4; t++) oacc[s2][t] = (v8f){};

    #pragma unroll 1
    for (int kb = 0; kb < Nn; kb += 32) {
        // scores: s1 + s2, two 16x16 tiles (cols kb..kb+15 and kb+16..kb+31)
        v8f slo = {}, shi = {};
        #pragma unroll
        for (int s2 = 0; s2 < 2; s2++) {
            const __bf16* kbp = Kd + s2 * SQ + qkbase;
            #pragma unroll
            for (int kc = 0; kc < 2; kc++) {
                v16bf blo = *(const v16bf*)(kbp + (size_t)(kb + l16) * Dd + kc * 32 + 16 * hv);
                v16bf bhi = *(const v16bf*)(kbp + (size_t)(kb + 16 + l16) * Dd + kc * 32 + 16 * hv);
                slo = wmma_bf16(aq[s2][kc], blo, slo);
                shi = wmma_bf16(aq[s2][kc], bhi, shi);
            }
        }
        // online softmax over the 32 new columns; row = 16 lanes of one half
        #pragma unroll
        for (int r = 0; r < 8; r++) {
            float mx = fmaxf(slo[r], shi[r]);
            mx = fmaxf(mx, __shfl_xor(mx, 1, 32));
            mx = fmaxf(mx, __shfl_xor(mx, 2, 32));
            mx = fmaxf(mx, __shfl_xor(mx, 4, 32));
            mx = fmaxf(mx, __shfl_xor(mx, 8, 32));
            float mnew  = fmaxf(mrow[r], mx);
            float alpha = __expf(mrow[r] - mnew);
            float plo   = __expf(slo[r] - mnew);
            float phi   = __expf(shi[r] - mnew);
            float sm = plo + phi;
            sm += __shfl_xor(sm, 1, 32);
            sm += __shfl_xor(sm, 2, 32);
            sm += __shfl_xor(sm, 4, 32);
            sm += __shfl_xor(sm, 8, 32);
            lrow[r] = lrow[r] * alpha + sm;
            mrow[r] = mnew;
            #pragma unroll
            for (int t = 0; t < 4; t++) { oacc[0][t][r] *= alpha; oacc[1][t][r] *= alpha; }
            pbuf[w][r + 8 * hv][l16]      = f2bf(plo);
            pbuf[w][r + 8 * hv][16 + l16] = f2bf(phi);
        }
        asm volatile("s_wait_dscnt 0" ::: "memory");   // LDS RAW within wave
        // re-layout P (D-format) -> A-format via LDS
        const __bf16* prow = &pbuf[w][l16][0];
        v16bf pa = *(const v16bf*)(prow + 16 * hv);
        // O += P @ V (both streams, 4 d-tiles)
        #pragma unroll
        for (int s2 = 0; s2 < 2; s2++) {
            const __bf16* vbase = Vt + s2 * SQ + (size_t)bh * Dd * Nn;
            #pragma unroll
            for (int t = 0; t < 4; t++) {
                v16bf bv = *(const v16bf*)(vbase + (size_t)(t * 16 + l16) * Nn + kb + 16 * hv);
                oacc[s2][t] = wmma_bf16(pa, bv, oacc[s2][t]);
            }
        }
    }

    const int bb = bh / Hh, hh = bh % Hh;
    #pragma unroll
    for (int r = 0; r < 8; r++) {
        const float inv = 1.0f / lrow[r];
        const int ntok = qt * 16 + r + 8 * hv;
        const size_t base = ((size_t)(bb * Nn + ntok)) * Cc + hh * Dd + l16;
        #pragma unroll
        for (int s2 = 0; s2 < 2; s2++)
            #pragma unroll
            for (int t = 0; t < 4; t++)
                O[s2 * BNC + base + t * 16] = f2bf(oacc[s2][t][r] * inv);
    }
}

// ---------------- kernel 3: output projection ----------------
// 32x32 tile per wave (2x2 register-blocked WMMA).
// Ob: [2][B*N][C] bf16, wp: [C][C] bf16, bias: [C] f32, out: [2][B*N][C] f32
__global__ void __launch_bounds__(128) proj_kernel(
    const __bf16* __restrict__ Ob, const __bf16* __restrict__ wp,
    const float* __restrict__ bias, float* __restrict__ out)
{
    constexpr int NT = Cc / 32;        // 24
    constexpr int MT = (Bb * Nn) / 32; // 128
    const int lane = threadIdx.x & 31;
    const int l16 = lane & 15, hv = lane >> 4;
    int wid = blockIdx.x * 4 + (threadIdx.x >> 5);
    const int s  = wid / (MT * NT); wid -= s * (MT * NT);
    const int mt = wid / NT;
    const int nt = wid - mt * NT;

    const __bf16* xrow0 = Ob + (size_t)s * BNC + (size_t)(mt * 32 + l16) * Cc;
    const __bf16* xrow1 = xrow0 + (size_t)16 * Cc;
    const __bf16* wrow0 = wp + (size_t)(nt * 32 + l16) * Cc + 16 * hv;
    const __bf16* wrow1 = wrow0 + (size_t)16 * Cc;

    v8f acc[2][2];
    acc[0][0] = (v8f){}; acc[0][1] = (v8f){}; acc[1][0] = (v8f){}; acc[1][1] = (v8f){};
    #pragma unroll 2
    for (int k0 = 0; k0 < Cc; k0 += 32) {
        v16bf a0 = *(const v16bf*)(xrow0 + k0 + 16 * hv);
        v16bf a1 = *(const v16bf*)(xrow1 + k0 + 16 * hv);
        v16bf b0 = *(const v16bf*)(wrow0 + k0);
        v16bf b1 = *(const v16bf*)(wrow1 + k0);
        acc[0][0] = wmma_bf16(a0, b0, acc[0][0]);
        acc[0][1] = wmma_bf16(a0, b1, acc[0][1]);
        acc[1][0] = wmma_bf16(a1, b0, acc[1][0]);
        acc[1][1] = wmma_bf16(a1, b1, acc[1][1]);
    }

    #pragma unroll
    for (int ns = 0; ns < 2; ns++) {
        const int n = nt * 32 + ns * 16 + l16;
        const float bval = bias[n];
        #pragma unroll
        for (int ms = 0; ms < 2; ms++) {
            #pragma unroll
            for (int r = 0; r < 8; r++) {
                const int m = mt * 32 + ms * 16 + r + 8 * hv;
                out[(size_t)s * BNC + (size_t)m * Cc + n] = acc[ms][ns][r] + bval;
            }
        }
    }
}

// ---------------- launcher ----------------
extern "C" void kernel_launch(void* const* d_in, const int* in_sizes, int n_in,
                              void* d_out, int out_size, void* d_ws, size_t ws_size,
                              hipStream_t stream) {
    const float* x1    = (const float*)d_in[0];
    const float* x2    = (const float*)d_in[1];
    const float* Wqkv  = (const float*)d_in[2];
    const float* bqkv  = (const float*)d_in[3];
    const float* Wproj = (const float*)d_in[4];
    const float* bproj = (const float*)d_in[5];
    float* out = (float*)d_out;

    __bf16* ws  = (__bf16*)d_ws;
    __bf16* xb  = ws;                               // 2*BNC
    __bf16* wqb = xb  + 2 * BNC;                    // 3*C*C
    __bf16* wpb = wqb + (size_t)3 * Cc * Cc;        // C*C
    __bf16* Qb  = wpb + (size_t)Cc * Cc;            // 2*BNC
    __bf16* Kb  = Qb  + 2 * BNC;                    // 2*BNC
    __bf16* Vtb = Kb  + 2 * BNC;                    // 2*BNC
    __bf16* Obf = Vtb + 2 * BNC;                    // 2*BNC

    cvt_kernel<<<1024, 256, 0, stream>>>(x1, xb, (int)BNC);
    cvt_kernel<<<1024, 256, 0, stream>>>(x2, xb + BNC, (int)BNC);
    cvt_kernel<<<512,  256, 0, stream>>>(Wqkv, wqb, 3 * Cc * Cc);
    cvt_kernel<<<256,  256, 0, stream>>>(Wproj, wpb, Cc * Cc);

    // 2 streams * 128 m-tiles * 72 n-tiles = 18432 waves / 4 per block
    qkv_kernel<<<4608, 128, 0, stream>>>(xb, wqb, bqkv, Qb, Kb, Vtb);
    // 48 (b,h) * 64 q-tiles = 3072 waves / 4 per block
    attn_kernel<<<768, 128, 0, stream>>>(Qb, Kb, Vtb, Obf);
    // 2 streams * 128 * 24 = 6144 waves / 4 per block
    proj_kernel<<<1536, 128, 0, stream>>>(Obf, wpb, bproj, out);
}
